// MGHD_13486197310275
// MI455X (gfx1250) — compile-verified, matrix operand order checked by reading, not run on hardware
//
#include <hip/hip_runtime.h>
#include <hip/hip_bf16.h>
#include <math.h>

// ---------------- model constants (from reference) ----------------
#define DISTC      25
#define N_IN       16
#define N_HID      128
#define N_OUT      2
#define D_MODEL    256
#define D_STATE    64
#define D_CONV     4
#define HEADDIM    64
#define D_INNER    512            // EXPAND * D_MODEL
#define NHEADS     8              // D_INNER / HEADDIM
#define CONV_DIM   640            // D_INNER + 2*D_STATE
#define D_IN_PROJ  1160           // 2*D_INNER + 2*D_STATE + NHEADS
#define NUM_CHECK  624            // DIST^2 - 1
#define NPG        1249           // NUM_CHECK + DIST^2
#define BATCH      64
#define N_NODES    79936          // BATCH * NPG
#define N_EDGES    639488         // N_NODES * 8
#define MTOK       39936          // BATCH * NUM_CHECK  (sequence tokens)
#define LD1        1216           // D_IN_PROJ padded to 19*64 for WMMA N-tiling

typedef __attribute__((ext_vector_type(16))) __bf16 v16bf;
typedef __attribute__((ext_vector_type(8)))  __bf16 v8bf;
typedef __attribute__((ext_vector_type(8)))  float  v8f;

__device__ __forceinline__ float siluf(float x) { return x / (1.0f + __expf(-x)); }

// Load one 16x32-bf16 WMMA fragment slice for this lane: two contiguous 16B chunks
// (K = kb..kb+7 and kb+16..kb+23 per the CDNA5 16-bit A/B lane layout).
__device__ __forceinline__ v16bf load_frag(const __bf16* __restrict__ p) {
    v8bf lo = *(const v8bf*)(p);
    v8bf hi = *(const v8bf*)(p + 16);
    return __builtin_shufflevector(lo, hi, 0, 1, 2, 3, 4, 5, 6, 7,
                                   8, 9, 10, 11, 12, 13, 14, 15);
}

// ---------------- K0: gather check nodes + embed (16 -> 256), write bf16 A ----------------
__global__ __launch_bounds__(256) void embed_kernel(const float* __restrict__ node_inputs,
                                                    const float* __restrict__ W_embed,
                                                    const float* __restrict__ b_embed,
                                                    __bf16* __restrict__ embA) {
    int i = blockIdx.x;            // token 0..MTOK-1
    int j = threadIdx.x;           // 0..255
    int b = i / NUM_CHECK;
    int t = i - b * NUM_CHECK;
    const float* x = node_inputs + (size_t)(b * NPG + t) * N_IN;
    float s = b_embed[j];
#pragma unroll
    for (int k = 0; k < N_IN; ++k) s += x[k] * W_embed[k * D_MODEL + j];
    embA[(size_t)i * D_MODEL + j] = (__bf16)s;
}

// ---------------- K1: weight prep (transpose to N-major bf16, pad, pad bias) ----------------
__global__ __launch_bounds__(256) void prep_weights(const float* __restrict__ W_in,
                                                    const float* __restrict__ b_in,
                                                    const float* __restrict__ W_outp,
                                                    __bf16* __restrict__ Wt1,    // [LD1][256]
                                                    float*  __restrict__ bias1,  // [LD1]
                                                    __bf16* __restrict__ Wt2) {  // [256][512]
    int idx = blockIdx.x * 256 + threadIdx.x;
    if (idx < LD1 * D_MODEL) {
        int n = idx / D_MODEL, k = idx - n * D_MODEL;
        float v = (n < D_IN_PROJ) ? W_in[(size_t)k * D_IN_PROJ + n] : 0.0f;
        Wt1[(size_t)n * D_MODEL + k] = (__bf16)v;
    }
    if (idx < LD1) bias1[idx] = (idx < D_IN_PROJ) ? b_in[idx] : 0.0f;
    if (idx < D_MODEL * D_INNER) {
        int n = idx / D_INNER, k = idx - n * D_INNER;
        Wt2[(size_t)n * D_INNER + k] = (__bf16)W_outp[(size_t)k * D_MODEL + n];
    }
}

// ---------------- WMMA bf16 GEMM: C[M,N] = A[M,K] @ Bt[N,K]^T + bias[N] ----------------
// block = 256 threads (8 waves); block tile = 128(M) x 64(N); wave tile = 16 x 64 (4 WMMA accums).
// Software-pipelined: fragments for step k+32 are issued before the WMMAs of step k, so
// global_load_b128s stay in flight across the matrix ops instead of s_wait_loadcnt 0 serializing.
// M % 128 == 0, N % 64 == 0, K % 32 == 0 (caller guarantees via padding).
__global__ __launch_bounds__(256) void wmma_gemm_bf16(const __bf16* __restrict__ A,
                                                      const __bf16* __restrict__ Bt,
                                                      const float*  __restrict__ bias,
                                                      float* __restrict__ C,
                                                      int M, int N, int K) {
    int wave  = threadIdx.x >> 5;
    int lane  = threadIdx.x & 31;
    int l16   = lane & 15;
    int half  = lane >> 4;                  // 0: K 0..7/16..23, 1: K 8..15/24..31
    int mBase = blockIdx.x * 128 + wave * 16;
    int nBase = blockIdx.y * 64;
    int kb    = half * 8;

    const __bf16* arow = A + (size_t)(mBase + l16) * K + kb;
    const __bf16* bcol0 = Bt + (size_t)(nBase + l16) * K + kb;
    const __bf16* bcol1 = bcol0 + (size_t)16 * K;
    const __bf16* bcol2 = bcol0 + (size_t)32 * K;
    const __bf16* bcol3 = bcol0 + (size_t)48 * K;

    v8f acc[4];
#pragma unroll
    for (int t = 0; t < 4; ++t) {
        float bv = bias[nBase + t * 16 + l16];
#pragma unroll
        for (int r = 0; r < 8; ++r) acc[t][r] = bv;
    }

    // prologue: fragments for k0 = 0
    v16bf af  = load_frag(arow);
    v16bf bf0 = load_frag(bcol0);
    v16bf bf1 = load_frag(bcol1);
    v16bf bf2 = load_frag(bcol2);
    v16bf bf3 = load_frag(bcol3);

    int k0 = 0;
    for (; k0 < K - 32; k0 += 32) {
        // issue next-step loads first (independent of the WMMAs below)
        v16bf an  = load_frag(arow  + k0 + 32);
        v16bf bn0 = load_frag(bcol0 + k0 + 32);
        v16bf bn1 = load_frag(bcol1 + k0 + 32);
        v16bf bn2 = load_frag(bcol2 + k0 + 32);
        v16bf bn3 = load_frag(bcol3 + k0 + 32);

        acc[0] = __builtin_amdgcn_wmma_f32_16x16x32_bf16(false, af, false, bf0,
                                                         (short)0, acc[0], false, false);
        acc[1] = __builtin_amdgcn_wmma_f32_16x16x32_bf16(false, af, false, bf1,
                                                         (short)0, acc[1], false, false);
        acc[2] = __builtin_amdgcn_wmma_f32_16x16x32_bf16(false, af, false, bf2,
                                                         (short)0, acc[2], false, false);
        acc[3] = __builtin_amdgcn_wmma_f32_16x16x32_bf16(false, af, false, bf3,
                                                         (short)0, acc[3], false, false);

        af = an; bf0 = bn0; bf1 = bn1; bf2 = bn2; bf3 = bn3;
    }
    // epilogue: last K-step
    acc[0] = __builtin_amdgcn_wmma_f32_16x16x32_bf16(false, af, false, bf0,
                                                     (short)0, acc[0], false, false);
    acc[1] = __builtin_amdgcn_wmma_f32_16x16x32_bf16(false, af, false, bf1,
                                                     (short)0, acc[1], false, false);
    acc[2] = __builtin_amdgcn_wmma_f32_16x16x32_bf16(false, af, false, bf2,
                                                     (short)0, acc[2], false, false);
    acc[3] = __builtin_amdgcn_wmma_f32_16x16x32_bf16(false, af, false, bf3,
                                                     (short)0, acc[3], false, false);

    // C/D layout: VGPR r: lanes 0-15 -> M=r, lanes 16-31 -> M=8+r; N = nBase + t*16 + (lane&15)
    int rBase = mBase + half * 8;
#pragma unroll
    for (int t = 0; t < 4; ++t) {
        int col = nBase + t * 16 + l16;
#pragma unroll
        for (int r = 0; r < 8; ++r)
            C[(size_t)(rBase + r) * N + col] = acc[t][r];
    }
}

// ---------------- K3: causal depthwise conv (width 4) + SiLU over xBC channels ----------------
__global__ __launch_bounds__(256) void conv_silu_kernel(const float* __restrict__ zx,
                                                        const float* __restrict__ conv_w,
                                                        const float* __restrict__ conv_b,
                                                        float* __restrict__ xBC) {
    size_t idx = (size_t)blockIdx.x * 256 + threadIdx.x;   // MTOK * CONV_DIM
    if (idx >= (size_t)MTOK * CONV_DIM) return;
    int c = (int)(idx % CONV_DIM);
    int i = (int)(idx / CONV_DIM);
    int b = i / NUM_CHECK, t = i - b * NUM_CHECK;
    float s = conv_b[c];
#pragma unroll
    for (int k = 0; k < D_CONV; ++k) {
        int tt = t - (D_CONV - 1) + k;
        if (tt >= 0)
            s += conv_w[c * D_CONV + k] * zx[(size_t)(b * NUM_CHECK + tt) * LD1 + D_INNER + c];
    }
    xBC[idx] = siluf(s);
}

// ---------------- K4: dt = softplus(dt + dt_bias); dA = exp(-exp(A_log)*dt) ----------------
__global__ __launch_bounds__(256) void dt_kernel(const float* __restrict__ zx,
                                                 const float* __restrict__ A_log,
                                                 const float* __restrict__ dt_bias,
                                                 float* __restrict__ dtb,
                                                 float* __restrict__ dAb) {
    int idx = blockIdx.x * 256 + threadIdx.x;              // MTOK * NHEADS
    if (idx >= MTOK * NHEADS) return;
    int h = idx & (NHEADS - 1);
    int i = idx >> 3;
    float x = zx[(size_t)i * LD1 + (D_INNER + CONV_DIM) + h] + dt_bias[h];
    float sp = (x > 20.0f) ? x : log1pf(__expf(x));
    dtb[idx] = sp;
    dAb[idx] = __expf(-__expf(A_log[h]) * sp);
}

// ---------------- K5: selective scan; one block per (batch, head); state in registers ---------
__global__ __launch_bounds__(256) void scan_kernel(const float* __restrict__ xBC,
                                                   const float* __restrict__ dtb,
                                                   const float* __restrict__ dAb,
                                                   const float* __restrict__ D_skip,
                                                   float* __restrict__ ybuf) {
    int b  = blockIdx.x >> 3;
    int hh = blockIdx.x & 7;
    int tid = threadIdx.x;          // 256
    int p   = tid & 63;             // head-dim element
    int n0  = (tid >> 6) * 16;      // 16-wide state slice

    float hreg[16];
#pragma unroll
    for (int j = 0; j < 16; ++j) hreg[j] = 0.0f;

    __shared__ float shx[64], shB[64], shC[64], shs[2];
    __shared__ float part[256];
    float dsk = D_skip[hh];

    for (int ts = 0; ts < NUM_CHECK; ++ts) {
        size_t row = (size_t)(b * NUM_CHECK + ts);
        if (tid < 64)            shx[tid]       = xBC[row * CONV_DIM + hh * HEADDIM + tid];
        else if (tid < 128)      shB[tid - 64]  = xBC[row * CONV_DIM + D_INNER + (tid - 64)];
        else if (tid < 192)      shC[tid - 128] = xBC[row * CONV_DIM + D_INNER + D_STATE + (tid - 128)];
        else if (tid == 192) { shs[0] = dAb[row * NHEADS + hh]; shs[1] = dtb[row * NHEADS + hh]; }
        __syncthreads();

        float dA   = shs[0];
        float coef = shs[1] * shx[p];
        float acc  = 0.0f;
#pragma unroll
        for (int j = 0; j < 16; ++j) {
            hreg[j] = dA * hreg[j] + coef * shB[n0 + j];
            acc += hreg[j] * shC[n0 + j];
        }
        part[tid] = acc;
        __syncthreads();
        if (tid < 64) {
            float y = part[tid] + part[tid + 64] + part[tid + 128] + part[tid + 192]
                    + dsk * shx[tid];
            ybuf[row * D_INNER + hh * HEADDIM + tid] = y;
        }
        __syncthreads();
    }
}

// ---------------- K6: gate with silu(z), RMSNorm, emit bf16 for out_proj GEMM ----------------
__global__ __launch_bounds__(256) void gate_norm_kernel(const float* __restrict__ ybuf,
                                                        const float* __restrict__ zx,
                                                        const float* __restrict__ norm_w,
                                                        __bf16* __restrict__ ybf) {
    int i = blockIdx.x;             // row 0..MTOK-1
    int tid = threadIdx.x;          // 256; 2 channels each
    __shared__ float red[256];
    float v[2];
    float ss = 0.0f;
#pragma unroll
    for (int u = 0; u < 2; ++u) {
        int d = tid + u * 256;
        float z = zx[(size_t)i * LD1 + d];
        float yv = ybuf[(size_t)i * D_INNER + d] * siluf(z);
        v[u] = yv;
        ss += yv * yv;
    }
    red[tid] = ss;
    __syncthreads();
    for (int s = 128; s > 0; s >>= 1) {
        if (tid < s) red[tid] += red[tid + s];
        __syncthreads();
    }
    float rs = rsqrtf(red[0] * (1.0f / D_INNER) + 1e-5f);
#pragma unroll
    for (int u = 0; u < 2; ++u) {
        int d = tid + u * 256;
        ybf[(size_t)i * D_INNER + d] = (__bf16)(v[u] * rs * norm_w[d]);
    }
}

// ---------------- zero fill ----------------
__global__ __launch_bounds__(256) void zero_kernel(float* __restrict__ p, size_t n) {
    size_t i = (size_t)blockIdx.x * 256 + threadIdx.x;
    if (i < n) p[i] = 0.0f;
}

// ---------------- K7: proj (256 -> 16) + scatter into feat at check-node slots ----------------
__global__ __launch_bounds__(256) void proj_scatter_kernel(const float* __restrict__ outm,
                                                           const float* __restrict__ W_proj,
                                                           const float* __restrict__ b_proj,
                                                           float* __restrict__ feat) {
    int gid = blockIdx.x * 256 + threadIdx.x;              // MTOK * 16
    if (gid >= MTOK * N_IN) return;
    int j = gid & 15;
    int i = gid >> 4;
    const float* r = outm + (size_t)i * D_MODEL;
    float s = b_proj[j];
    for (int k = 0; k < D_MODEL; ++k) s += r[k] * W_proj[k * N_IN + j];
    int b = i / NUM_CHECK, t = i - b * NUM_CHECK;
    feat[(size_t)(b * NPG + t) * N_IN + j] = s;
}

// ---------------- K8: edge message (16->128, relu) fused with atomic segment-sum ----------------
__global__ __launch_bounds__(256) void msg_agg_kernel(const float* __restrict__ feat,
                                                      const int* __restrict__ src,
                                                      const int* __restrict__ dst,
                                                      const float* __restrict__ W_msg,
                                                      const float* __restrict__ b_msg,
                                                      float* __restrict__ agg) {
    size_t gid = (size_t)blockIdx.x * 256 + threadIdx.x;   // N_EDGES * 128
    if (gid >= (size_t)N_EDGES * N_HID) return;
    int j = (int)(gid & (N_HID - 1));
    size_t e = gid >> 7;
    int s = src[e], d = dst[e];
    const float* f = feat + (size_t)s * N_IN;
    float m = b_msg[j];
#pragma unroll
    for (int k = 0; k < N_IN; ++k) m += f[k] * W_msg[k * N_HID + j];
    m = fmaxf(m, 0.0f);
    atomicAdd(&agg[(size_t)d * N_HID + j], m);
}

// ---------------- K9: node update MLP (144 -> 128, relu) ----------------
__global__ __launch_bounds__(256) void upd_kernel(const float* __restrict__ feat,
                                                  const float* __restrict__ agg,
                                                  const float* __restrict__ W_upd,
                                                  const float* __restrict__ b_upd,
                                                  float* __restrict__ hbuf) {
    size_t gid = (size_t)blockIdx.x * 256 + threadIdx.x;   // N_NODES * 128
    if (gid >= (size_t)N_NODES * N_HID) return;
    int j = (int)(gid & (N_HID - 1));
    size_t n = gid >> 7;
    float s = b_upd[j];
    const float* f = feat + n * N_IN;
#pragma unroll
    for (int k = 0; k < N_IN; ++k) s += f[k] * W_upd[(size_t)k * N_HID + j];
    const float* a = agg + n * N_HID;
    for (int k = 0; k < N_HID; ++k) s += a[k] * W_upd[(size_t)(N_IN + k) * N_HID + j];
    hbuf[gid] = fmaxf(s, 0.0f);
}

// ---------------- K10: output head (128 -> 2) ----------------
__global__ __launch_bounds__(256) void out_kernel(const float* __restrict__ hbuf,
                                                  const float* __restrict__ W_out,
                                                  const float* __restrict__ b_out,
                                                  float* __restrict__ out) {
    size_t gid = (size_t)blockIdx.x * 256 + threadIdx.x;   // N_NODES * 2
    if (gid >= (size_t)N_NODES * N_OUT) return;
    int j = (int)(gid & 1);
    size_t n = gid >> 1;
    const float* hr = hbuf + n * N_HID;
    float s = b_out[j];
    for (int k = 0; k < N_HID; ++k) s += hr[k] * W_out[k * N_OUT + j];
    out[gid] = s;
}

// =======================================================================================
extern "C" void kernel_launch(void* const* d_in, const int* in_sizes, int n_in,
                              void* d_out, int out_size, void* d_ws, size_t ws_size,
                              hipStream_t stream) {
    (void)in_sizes; (void)n_in; (void)out_size; (void)ws_size;
    const float* node_inputs = (const float*)d_in[0];
    const int*   src_ids     = (const int*)d_in[1];
    const int*   dst_ids     = (const int*)d_in[2];
    const float* W_embed     = (const float*)d_in[3];
    const float* b_embed     = (const float*)d_in[4];
    const float* W_in        = (const float*)d_in[5];
    const float* b_in        = (const float*)d_in[6];
    const float* conv_w      = (const float*)d_in[7];
    const float* conv_b      = (const float*)d_in[8];
    const float* A_log       = (const float*)d_in[9];
    const float* dt_bias     = (const float*)d_in[10];
    const float* D_skip      = (const float*)d_in[11];
    const float* norm_w      = (const float*)d_in[12];
    const float* W_outp      = (const float*)d_in[13];
    const float* b_outp      = (const float*)d_in[14];
    const float* W_proj      = (const float*)d_in[15];
    const float* b_proj      = (const float*)d_in[16];
    const float* W_msg       = (const float*)d_in[17];
    const float* b_msg       = (const float*)d_in[18];
    const float* W_upd       = (const float*)d_in[19];
    const float* b_upd       = (const float*)d_in[20];
    const float* W_out       = (const float*)d_in[21];
    const float* b_out       = (const float*)d_in[22];
    float* out = (float*)d_out;

    // ---- workspace carving (256B aligned slabs) ----
    char* ws = (char*)d_ws;
    size_t off = 0;
    auto carve = [&](size_t bytes) -> char* {
        char* p = ws + off;
        off += (bytes + 255) & ~(size_t)255;
        return p;
    };
    __bf16* embA  = (__bf16*)carve((size_t)MTOK * D_MODEL * 2);        // 20.4 MB
    __bf16* Wt1   = (__bf16*)carve((size_t)LD1 * D_MODEL * 2);         // 0.62 MB
    float*  bias1 = (float*)carve((size_t)LD1 * 4);
    __bf16* Wt2   = (__bf16*)carve((size_t)D_MODEL * D_INNER * 2);     // 0.26 MB
    float*  zx    = (float*)carve((size_t)MTOK * LD1 * 4);             // 194 MB
    float*  xBC   = (float*)carve((size_t)MTOK * CONV_DIM * 4);        // 102 MB
    float*  dtb   = (float*)carve((size_t)MTOK * NHEADS * 4);
    float*  dAb   = (float*)carve((size_t)MTOK * NHEADS * 4);
    float*  ybuf  = (float*)carve((size_t)MTOK * D_INNER * 4);         // 82 MB
    __bf16* ybf   = (__bf16*)carve((size_t)MTOK * D_INNER * 2);        // 41 MB
    float*  outm  = (float*)carve((size_t)MTOK * D_MODEL * 4);         // 41 MB
    float*  feat  = (float*)carve((size_t)N_NODES * N_IN * 4);         // 5.1 MB
    float*  agg   = (float*)carve((size_t)N_NODES * N_HID * 4);        // 41 MB
    float*  hbuf  = (float*)carve((size_t)N_NODES * N_HID * 4);        // 41 MB

    dim3 blk(256);

    // K0: embed (gather + 16x256 matvec) -> bf16 A matrix
    embed_kernel<<<dim3(MTOK), blk, 0, stream>>>(node_inputs, W_embed, b_embed, embA);

    // K1: weight prep
    prep_weights<<<dim3((LD1 * D_MODEL + 255) / 256), blk, 0, stream>>>(
        W_in, b_in, W_outp, Wt1, bias1, Wt2);

    // K2: in_proj GEMM  zx[M,1216] = embA[M,256] @ Wt1^T + bias1   (WMMA bf16)
    wmma_gemm_bf16<<<dim3(MTOK / 128, LD1 / 64), blk, 0, stream>>>(
        embA, Wt1, bias1, zx, MTOK, LD1, D_MODEL);

    // K3: causal depthwise conv + SiLU
    {
        size_t n = (size_t)MTOK * CONV_DIM;
        conv_silu_kernel<<<dim3((unsigned)((n + 255) / 256)), blk, 0, stream>>>(
            zx, conv_w, conv_b, xBC);
    }

    // K4: dt / dA precompute
    dt_kernel<<<dim3((MTOK * NHEADS + 255) / 256), blk, 0, stream>>>(
        zx, A_log, dt_bias, dtb, dAb);

    // K5: selective scan (512 blocks = batch x head, state resident in VGPRs)
    scan_kernel<<<dim3(BATCH * NHEADS), blk, 0, stream>>>(xBC, dtb, dAb, D_skip, ybuf);

    // K6: gate + RMSNorm -> bf16
    gate_norm_kernel<<<dim3(MTOK), blk, 0, stream>>>(ybuf, zx, norm_w, ybf);

    // K7: out_proj GEMM  outm[M,256] = ybf[M,512] @ Wt2^T + b_outp  (WMMA bf16)
    wmma_gemm_bf16<<<dim3(MTOK / 128, D_MODEL / 64), blk, 0, stream>>>(
        ybf, Wt2, b_outp, outm, MTOK, D_MODEL, D_INNER);

    // zero feat + agg (harness poisons ws; must re-zero every call)
    {
        size_t nf = (size_t)N_NODES * N_IN;
        zero_kernel<<<dim3((unsigned)((nf + 255) / 256)), blk, 0, stream>>>(feat, nf);
        size_t na = (size_t)N_NODES * N_HID;
        zero_kernel<<<dim3((unsigned)((na + 255) / 256)), blk, 0, stream>>>(agg, na);
    }

    // K8: proj + scatter
    proj_scatter_kernel<<<dim3((MTOK * N_IN + 255) / 256), blk, 0, stream>>>(
        outm, W_proj, b_proj, feat);

    // K9: edge messages + atomic segment-sum
    {
        size_t n = (size_t)N_EDGES * N_HID;
        msg_agg_kernel<<<dim3((unsigned)((n + 255) / 256)), blk, 0, stream>>>(
            feat, src_ids, dst_ids, W_msg, b_msg, agg);
    }

    // K10: node update MLP
    {
        size_t n = (size_t)N_NODES * N_HID;
        upd_kernel<<<dim3((unsigned)((n + 255) / 256)), blk, 0, stream>>>(
            feat, agg, W_upd, b_upd, hbuf);
    }

    // K11: output head
    {
        size_t n = (size_t)N_NODES * N_OUT;
        out_kernel<<<dim3((unsigned)((n + 255) / 256)), blk, 0, stream>>>(
            hbuf, W_out, b_out, out);
    }
}
